// Model_49718541418915
// MI455X (gfx1250) — matrix operand order repaired and checked by reading, not compiled, not run on hardware
//
#include <hip/hip_runtime.h>
#include <hip/hip_bf16.h>

typedef _Float16 f16;
typedef __attribute__((ext_vector_type(16))) _Float16 v16h;
typedef __attribute__((ext_vector_type(8)))  _Float16 v8h;
typedef __attribute__((ext_vector_type(8)))  float    v8f;

#define N_NODES 100000
#define N_EDGES 3200000
#define NFEAT 128
#define HID 32
#define NPG 200
#define NGRAPH 500
#define KPOOL 50

// ---------------- small utility kernels ----------------

__global__ void k_zero_f32(float* p, int n) {
  int t = blockIdx.x * blockDim.x + threadIdx.x;
  if (t < n) p[t] = 0.f;
}
__global__ void k_zero_f16(f16* p, int n) {
  int t = blockIdx.x * blockDim.x + threadIdx.x;
  if (t < n) p[t] = (f16)0.f;
}
__global__ void k_cvt_f16(const float* __restrict__ s, f16* __restrict__ d, int n) {
  int t = blockIdx.x * blockDim.x + threadIdx.x;
  if (t < n) d[t] = (f16)s[t];
}
// W [K][N] f32 -> Wt [N][K] f16
__global__ void k_transpose_f16(const float* __restrict__ W, f16* __restrict__ Wt, int K, int N) {
  int t = blockIdx.x * blockDim.x + threadIdx.x;
  if (t < K * N) {
    int k = t / N, n = t % N;
    Wt[n * K + k] = (f16)W[t];
  }
}

// ---------------- degree / normalization ----------------

__global__ void k_deg_count(const int* __restrict__ ei, float* __restrict__ deg) {
  int e = blockIdx.x * blockDim.x + threadIdx.x;
  if (e >= N_EDGES) return;
  int s = ei[e], d = ei[N_EDGES + e];
  if (s != d) atomicAdd(&deg[d], 1.0f);
}
__global__ void k_make_dinv(float* deg) {
  int v = blockIdx.x * blockDim.x + threadIdx.x;
  if (v < N_NODES) deg[v] = rsqrtf(deg[v] + 1.0f);
}

// ---------------- WMMA fragment helpers ----------------
// A 16x32 f16: lanes 0-15 row M=lane, elems0-7=K kb+0..7, elems8-15=kb+16..23
//              lanes16-31 row M=lane-16, elems0-7=kb+8..15, elems8-15=kb+24..31
__device__ __forceinline__ v16h load_a(const f16* __restrict__ arow, int kb, int hi) {
  const f16* p = arow + kb + hi * 8;
  v8h lo = *(const v8h*)p;
  v8h hh = *(const v8h*)(p + 16);
  v16h a;
#pragma unroll
  for (int j = 0; j < 8; ++j) { a[j] = lo[j]; a[8 + j] = hh[j]; }
  return a;
}
// B 32x16 f16: lanes 0-15 col N=lane, elems0-7=K kb+0..7, elems8-15=kb+16..23; lanes16-31 col N=lane-16, elems0-7=kb+8..15, elems8-15=kb+24..31
__device__ __forceinline__ v16h load_b(const f16* __restrict__ brow, int kb, int hi) {
  const f16* p = brow + kb + hi * 8;
  v8h lo = *(const v8h*)p;
  v8h hh = *(const v8h*)(p + 16);
  v16h b;
#pragma unroll
  for (int j = 0; j < 8; ++j) { b[j] = lo[j]; b[8 + j] = hh[j]; }
  return b;
}

// ---------------- GCN matmul: C[M x 32] = A[M x K](f16) * Wt[32 x K]^T ----------------

__global__ void k_gemm_gcn(const f16* __restrict__ A, const f16* __restrict__ Wt,
                           float* __restrict__ C, int M, int K) {
  int mtile = blockIdx.x * blockDim.y + threadIdx.y;
  int ntile = blockIdx.y;  // 0..1 (N=32)
  if (mtile * 16 >= M) return;
  int lane = threadIdx.x, hi = lane >> 4;
  const f16* arow = A + (size_t)(mtile * 16 + (lane & 15)) * K;
  const f16* brow = Wt + (size_t)(ntile * 16 + (lane & 15)) * K;
  v8f acc = {};
  for (int kb = 0; kb < K; kb += 32) {
    v16h a = load_a(arow, kb, hi);
    v16h b = load_b(brow, kb, hi);
    acc = __builtin_amdgcn_wmma_f32_16x16x32_f16(false, a, false, b, (short)0, acc, false, false);
  }
  int n = ntile * 16 + (lane & 15);
  int mbase = mtile * 16 + hi * 8;
#pragma unroll
  for (int r = 0; r < 8; ++r) {
    int m = mbase + r;
    if (m < M) C[(size_t)m * 32 + n] = acc[r];
  }
}

// ---------------- edge aggregation (8 threads / edge, 4 channels each) ----------------

__global__ void k_edge_agg(const int* __restrict__ ei, const float* __restrict__ dinv,
                           const float* __restrict__ h, float* __restrict__ agg) {
  long tid = (long)blockIdx.x * blockDim.x + threadIdx.x;
  if (tid >= (long)N_EDGES * 8) return;
  int e = (int)(tid >> 3);
  int c0 = ((int)tid & 7) * 4;
  int s = ei[e], d = ei[N_EDGES + e];
  if (s == d) return;
  float nrm = dinv[s] * dinv[d];
  const float* hs = h + (size_t)s * 32 + c0;
  float* ad = agg + (size_t)d * 32 + c0;
#pragma unroll
  for (int j = 0; j < 4; ++j) atomicAdd(ad + j, nrm * hs[j]);
}

// ---------------- finalize: tanh(agg + h*dinv^2 + b); write xc col-slice + f16 copy ----------------

__global__ void k_finalize(const float* __restrict__ h, const float* __restrict__ agg,
                           const float* __restrict__ dinv, const float* __restrict__ bias,
                           float* __restrict__ xc, f16* __restrict__ hout, int colbase) {
  int t = blockIdx.x * blockDim.x + threadIdx.x;
  if (t >= N_NODES * 32) return;
  int v = t >> 5, c = t & 31;
  float di = dinv[v];
  float val = agg[t] + h[t] * di * di + bias[c];
  float r = tanhf(val);
  xc[(size_t)v * 96 + colbase + c] = r;
  hout[t] = (f16)r;
}

// ---------------- global sort pool: top-50 by channel 95, stable desc ----------------

__global__ void k_sort_pool(const float* __restrict__ xc, f16* __restrict__ pooled) {
  int b = blockIdx.x, tid = threadIdx.x;
  __shared__ float val[NPG];
  __shared__ short order[KPOOL];
  if (tid < NPG) val[tid] = xc[((size_t)b * NPG + tid) * 96 + 95];
  __syncthreads();
  if (tid < NPG) {
    float v = val[tid];
    int r = 0;
    for (int j = 0; j < NPG; ++j) {
      float vj = val[j];
      if (vj > v || (vj == v && j < tid)) ++r;
    }
    if (r < KPOOL) order[r] = (short)tid;
  }
  __syncthreads();
  for (int idx = tid; idx < KPOOL * 96; idx += 256) {
    int s = idx / 96, c = idx % 96;
    int node = order[s];
    pooled[(size_t)b * 4800 + idx] = (f16)xc[((size_t)b * NPG + node) * 96 + c];
  }
}

// ---------------- capsule 1 priors: 50 GEMMs of 512x512x96 ----------------
// U: [512][50][96] f16 ; Wc: [16][50][32][96] f16 ; P: [500][16][50][32] f32

__global__ void k_caps1_priors(const f16* __restrict__ U, const f16* __restrict__ Wc,
                               float* __restrict__ P) {
  int mtile = blockIdx.x;                                  // 0..31 (b)
  int ntile = blockIdx.y * blockDim.y + threadIdx.y;       // 0..31 (o,L)
  int i = blockIdx.z;                                      // 0..49
  int lane = threadIdx.x, hi = lane >> 4;
  int n = ntile * 16 + (lane & 15);
  int o = n >> 5, L = n & 31;
  const f16* arow = U + (size_t)(mtile * 16 + (lane & 15)) * 4800 + i * 96;
  const f16* brow = Wc + (size_t)o * 153600 + (size_t)i * 3072 + (size_t)L * 96;
  v8f acc = {};
#pragma unroll
  for (int kb = 0; kb < 96; kb += 32) {
    v16h a = load_a(arow, kb, hi);
    v16h b = load_b(brow, kb, hi);
    acc = __builtin_amdgcn_wmma_f32_16x16x32_f16(false, a, false, b, (short)0, acc, false, false);
  }
  int mbase = mtile * 16 + hi * 8;
#pragma unroll
  for (int r = 0; r < 8; ++r) {
    int m = mbase + r;
    if (m < NGRAPH) P[(size_t)m * 25600 + o * 1600 + i * 32 + L] = acc[r];
  }
}

// ---------------- routing for capsule layer 1 (O=16, I=50, L=32) ----------------

__global__ void k_routing1(const float* __restrict__ P, f16* __restrict__ Vout) {
  int b = blockIdx.x, tid = threadIdx.x;
  __shared__ float lg[800], cc[800], sv[512], vv[512];
  const float* Pb = P + (size_t)b * 25600;
  for (int k = tid; k < 800; k += 256) lg[k] = 0.f;
  __syncthreads();
  for (int it = 0; it < 3; ++it) {
    if (tid < 50) {
      int i = tid;
      float mx = -1e30f;
      for (int o = 0; o < 16; ++o) mx = fmaxf(mx, lg[o * 50 + i]);
      float sum = 0.f;
      for (int o = 0; o < 16; ++o) { float e = __expf(lg[o * 50 + i] - mx); cc[o * 50 + i] = e; sum += e; }
      float inv = 1.f / sum;
      for (int o = 0; o < 16; ++o) cc[o * 50 + i] *= inv;
    }
    __syncthreads();
    for (int ol = tid; ol < 512; ol += 256) {
      int o = ol >> 5, L = ol & 31;
      float acc = 0.f;
      const float* pp = Pb + o * 1600 + L;
      for (int i = 0; i < 50; ++i) acc += cc[o * 50 + i] * pp[i * 32];
      sv[ol] = acc;
    }
    __syncthreads();
    if (tid < 16) {
      int o = tid;
      float sq = 0.f;
      for (int L = 0; L < 32; ++L) { float s = sv[o * 32 + L]; sq += s * s; }
      float f = (sq / (1.f + sq)) * rsqrtf(sq + 1e-12f);
      for (int L = 0; L < 32; ++L) vv[o * 32 + L] = sv[o * 32 + L] * f;
    }
    __syncthreads();
    if (it < 2) {
      for (int oi = tid; oi < 800; oi += 256) {
        int o = oi / 50, i = oi % 50;
        const float* pp = Pb + o * 1600 + i * 32;
        float dot = 0.f;
        for (int L = 0; L < 32; ++L) dot += pp[L] * vv[o * 32 + L];
        lg[oi] += dot;
      }
      __syncthreads();
    }
  }
  for (int k = tid; k < 512; k += 256) Vout[(size_t)b * 512 + k] = (f16)vv[k];
}

// ---------------- capsule 2 priors: 16 GEMMs of 512x160x32 ----------------
// U: [512][16][32] f16 ; Wc: [10][16][16][32] f16 ; P: [500][10][16][16] f32

__global__ void k_caps2_priors(const f16* __restrict__ U, const f16* __restrict__ Wc,
                               float* __restrict__ P) {
  int mtile = blockIdx.x;                                  // 0..31
  int ntile = blockIdx.y * blockDim.y + threadIdx.y;       // 0..9  (== o)
  int i = blockIdx.z;                                      // 0..15
  int lane = threadIdx.x, hi = lane >> 4;
  int o = ntile, L = lane & 15;
  const f16* arow = U + (size_t)(mtile * 16 + (lane & 15)) * 512 + i * 32;
  const f16* brow = Wc + (size_t)o * 8192 + i * 512 + (size_t)L * 32;
  v16h a = load_a(arow, 0, hi);
  v16h b = load_b(brow, 0, hi);
  v8f acc = {};
  acc = __builtin_amdgcn_wmma_f32_16x16x32_f16(false, a, false, b, (short)0, acc, false, false);
  int mbase = mtile * 16 + hi * 8;
#pragma unroll
  for (int r = 0; r < 8; ++r) {
    int m = mbase + r;
    if (m < NGRAPH) P[(size_t)m * 2560 + o * 256 + i * 16 + L] = acc[r];
  }
}

// ---------------- routing for capsule layer 2 (O=10, I=16, L=16) + final norm ----------------

__global__ void k_routing2(const float* __restrict__ P, float* __restrict__ Out) {
  int b = blockIdx.x, tid = threadIdx.x;
  __shared__ float lg[160], cc[160], sv[160], vv[160];
  const float* Pb = P + (size_t)b * 2560;
  if (tid < 160) lg[tid] = 0.f;
  __syncthreads();
  for (int it = 0; it < 3; ++it) {
    if (tid < 16) {
      int i = tid;
      float mx = -1e30f;
      for (int o = 0; o < 10; ++o) mx = fmaxf(mx, lg[o * 16 + i]);
      float sum = 0.f;
      for (int o = 0; o < 10; ++o) { float e = __expf(lg[o * 16 + i] - mx); cc[o * 16 + i] = e; sum += e; }
      float inv = 1.f / sum;
      for (int o = 0; o < 10; ++o) cc[o * 16 + i] *= inv;
    }
    __syncthreads();
    if (tid < 160) {
      int o = tid >> 4, L = tid & 15;
      float acc = 0.f;
      const float* pp = Pb + o * 256 + L;
      for (int i = 0; i < 16; ++i) acc += cc[o * 16 + i] * pp[i * 16];
      sv[tid] = acc;
    }
    __syncthreads();
    if (tid < 10) {
      int o = tid;
      float sq = 0.f;
      for (int L = 0; L < 16; ++L) { float s = sv[o * 16 + L]; sq += s * s; }
      float f = (sq / (1.f + sq)) * rsqrtf(sq + 1e-12f);
      for (int L = 0; L < 16; ++L) vv[o * 16 + L] = sv[o * 16 + L] * f;
    }
    __syncthreads();
    if (it < 2) {
      if (tid < 160) {
        int o = tid >> 4, i = tid & 15;
        const float* pp = Pb + o * 256 + i * 16;
        float dot = 0.f;
        for (int L = 0; L < 16; ++L) dot += pp[L] * vv[o * 16 + L];
        lg[tid] += dot;
      }
      __syncthreads();
    }
  }
  if (tid < 10) {
    float sq = 0.f;
    for (int L = 0; L < 16; ++L) { float v = vv[tid * 16 + L]; sq += v * v; }
    Out[(size_t)b * 10 + tid] = sqrtf(sq);
  }
}

// ---------------- launch ----------------

extern "C" void kernel_launch(void* const* d_in, const int* in_sizes, int n_in,
                              void* d_out, int out_size, void* d_ws, size_t ws_size,
                              hipStream_t stream) {
  (void)in_sizes; (void)n_in; (void)out_size; (void)ws_size;
  const float* x   = (const float*)d_in[0];
  const int*   ei  = (const int*)d_in[1];
  // d_in[2] = batch (implicit: 200 nodes per graph)
  const float* W1  = (const float*)d_in[3];
  const float* b1  = (const float*)d_in[4];
  const float* W2  = (const float*)d_in[5];
  const float* b2  = (const float*)d_in[6];
  const float* W3  = (const float*)d_in[7];
  const float* b3  = (const float*)d_in[8];
  const float* Wc1 = (const float*)d_in[9];
  const float* Wc2 = (const float*)d_in[10];
  float* out = (float*)d_out;

  char* ws = (char*)d_ws;
  size_t off = 0;
  auto take = [&](size_t bytes) { char* p = ws + off; off += (bytes + 255) & ~(size_t)255; return p; };
  f16*   x16    = (f16*)take((size_t)N_NODES * NFEAT * 2);
  f16*   w1t    = (f16*)take(32 * 128 * 2);
  f16*   w2t    = (f16*)take(32 * 32 * 2);
  f16*   w3t    = (f16*)take(32 * 32 * 2);
  f16*   wc1h   = (f16*)take((size_t)16 * 50 * 32 * 96 * 2);
  f16*   wc2h   = (f16*)take((size_t)10 * 16 * 16 * 32 * 2);
  float* dinv   = (float*)take((size_t)N_NODES * 4);
  float* h      = (float*)take((size_t)N_NODES * 32 * 4);
  float* agg    = (float*)take((size_t)N_NODES * 32 * 4);
  f16*   hin    = (f16*)take((size_t)N_NODES * 32 * 2);
  float* xc     = (float*)take((size_t)N_NODES * 96 * 4);
  f16*   pooled = (f16*)take((size_t)512 * 50 * 96 * 2);
  float* pr1    = (float*)take((size_t)NGRAPH * 16 * 50 * 32 * 4);
  f16*   v1h    = (f16*)take((size_t)512 * 16 * 32 * 2);
  float* pr2    = (float*)take((size_t)NGRAPH * 10 * 16 * 16 * 4);

  const int T = 256;
  // conversions / weight prep
  k_cvt_f16<<<(N_NODES * NFEAT + T - 1) / T, T, 0, stream>>>(x, x16, N_NODES * NFEAT);
  k_transpose_f16<<<(4096 + T - 1) / T, T, 0, stream>>>(W1, w1t, 128, 32);
  k_transpose_f16<<<(1024 + T - 1) / T, T, 0, stream>>>(W2, w2t, 32, 32);
  k_transpose_f16<<<(1024 + T - 1) / T, T, 0, stream>>>(W3, w3t, 32, 32);
  k_cvt_f16<<<(2457600 + T - 1) / T, T, 0, stream>>>(Wc1, wc1h, 2457600);
  k_cvt_f16<<<(81920 + T - 1) / T, T, 0, stream>>>(Wc2, wc2h, 81920);

  // degrees -> dinv (in place)
  k_zero_f32<<<(N_NODES + T - 1) / T, T, 0, stream>>>(dinv, N_NODES);
  k_deg_count<<<(N_EDGES + T - 1) / T, T, 0, stream>>>(ei, dinv);
  k_make_dinv<<<(N_NODES + T - 1) / T, T, 0, stream>>>(dinv);

  // 3 GCN layers
  dim3 gemmBlk(32, 4);
  dim3 gemmGrd((6250 + 3) / 4, 2);
  const f16* win[3] = {x16, hin, hin};
  const f16* wts[3] = {w1t, w2t, w3t};
  const float* bias[3] = {b1, b2, b3};
  const int kdim[3] = {128, 32, 32};
  for (int layer = 0; layer < 3; ++layer) {
    k_gemm_gcn<<<gemmGrd, gemmBlk, 0, stream>>>(win[layer], wts[layer], h, N_NODES, kdim[layer]);
    k_zero_f32<<<(N_NODES * 32 + T - 1) / T, T, 0, stream>>>(agg, N_NODES * 32);
    k_edge_agg<<<(int)(((long)N_EDGES * 8 + T - 1) / T), T, 0, stream>>>(ei, dinv, h, agg);
    k_finalize<<<(N_NODES * 32 + T - 1) / T, T, 0, stream>>>(h, agg, dinv, bias[layer], xc, hin,
                                                            layer * 32);
  }

  // sort pool -> pooled f16 (pad rows 500..511 with zeros)
  k_zero_f16<<<(12 * 4800 + T - 1) / T, T, 0, stream>>>(pooled + (size_t)500 * 4800, 12 * 4800);
  k_sort_pool<<<NGRAPH, T, 0, stream>>>(xc, pooled);

  // capsule layer 1
  k_caps1_priors<<<dim3(32, 8, 50), dim3(32, 4), 0, stream>>>(pooled, wc1h, pr1);
  k_zero_f16<<<(12 * 512 + T - 1) / T, T, 0, stream>>>(v1h + (size_t)500 * 512, 12 * 512);
  k_routing1<<<NGRAPH, T, 0, stream>>>(pr1, v1h);

  // capsule layer 2 + final norm
  k_caps2_priors<<<dim3(32, 5, 16), dim3(32, 2), 0, stream>>>(v1h, wc2h, pr2);
  k_routing2<<<NGRAPH, T, 0, stream>>>(pr2, out);
}
